// SparseFactorisationDense_44470091383360
// MI455X (gfx1250) — matrix operand order repaired and checked by reading, not compiled, not run on hardware
//
#include <hip/hip_runtime.h>
#include <stdint.h>

#define DD 2048   // K == N == feature dim
#define MM 4096   // batch rows

typedef __attribute__((ext_vector_type(16))) __bf16 v16bf;
typedef __attribute__((ext_vector_type(8)))  float  v8f;
typedef __attribute__((ext_vector_type(4)))  int    v4i;

#if __has_builtin(__builtin_amdgcn_global_load_async_to_lds_b128)
#define USE_ASYNC_LDS 1
typedef __attribute__((address_space(1))) v4i* gptr4;
typedef __attribute__((address_space(3))) v4i* lptr4;
#endif

__device__ __forceinline__ unsigned short f32_to_bf16_rne(float f) {
  unsigned int u = __float_as_uint(f);
  u += 0x7FFFu + ((u >> 16) & 1u);   // round-to-nearest-even
  return (unsigned short)(u >> 16);
}

// 16-byte global -> LDS copy: async (ASYNCcnt, no VGPR round-trip) if available
__device__ __forceinline__ void copy16(unsigned short* lds, const unsigned short* g) {
#ifdef USE_ASYNC_LDS
  __builtin_amdgcn_global_load_async_to_lds_b128((gptr4)g, (lptr4)lds, 0, 0);
#else
  *(uint4*)lds = *(const uint4*)g;
#endif
}

__device__ __forceinline__ void wait_async() {
#ifdef USE_ASYNC_LDS
#if __has_builtin(__builtin_amdgcn_s_wait_asynccnt)
  __builtin_amdgcn_s_wait_asynccnt(0);
#else
  asm volatile("s_wait_asynccnt 0x0" ::: "memory");
#endif
#endif
}

// ---------------------------------------------------------------------------
// x (f32, row-major [MM][DD]) -> bf16 same layout
// ---------------------------------------------------------------------------
__global__ __launch_bounds__(256) void cvt_x_kernel(const float* __restrict__ x,
                                                    unsigned short* __restrict__ xb) {
  int i = (blockIdx.x * 256 + threadIdx.x) * 4;
  float4 v = *(const float4*)(x + i);
  union { unsigned short s[4]; uint2 u; } p;
  p.s[0] = f32_to_bf16_rne(v.x);
  p.s[1] = f32_to_bf16_rne(v.y);
  p.s[2] = f32_to_bf16_rne(v.z);
  p.s[3] = f32_to_bf16_rne(v.w);
  *(uint2*)(xb + i) = p.u;
}

// ---------------------------------------------------------------------------
// W = k * mask (f32 [K][N]) -> bf16 TRANSPOSED  Wt[n][k]   (LDS tile transpose)
// ---------------------------------------------------------------------------
__global__ __launch_bounds__(256) void cvt_w_kernel(const float* __restrict__ k,
                                                    const int* __restrict__ m,
                                                    unsigned short* __restrict__ wt) {
  __shared__ unsigned short tile[32][33];   // +1 pad: avoid bank conflicts
  const int n0 = blockIdx.x * 32;
  const int k0 = blockIdx.y * 32;
  const int tx = threadIdx.x;               // 0..31
  const int ty = threadIdx.y;               // 0..7
  #pragma unroll
  for (int r = ty; r < 32; r += 8) {
    int kk = k0 + r, nn = n0 + tx;
    float w = k[(size_t)kk * DD + nn] * (float)m[(size_t)kk * DD + nn];
    tile[tx][r] = f32_to_bf16_rne(w);       // tile[n][k]
  }
  __syncthreads();
  #pragma unroll
  for (int r = ty; r < 32; r += 8) {
    wt[(size_t)(n0 + r) * DD + k0 + tx] = tile[r][tx];
  }
}

// ---------------------------------------------------------------------------
// C[M][N] = A[M][K] @ Wt[N][K]^T   (bf16 in, f32 accumulate)
//   block tile 128x256, BK = 32, 256 threads = 8 waves (2 x 4)
//   each wave: 64x64 = 4x4 WMMA accumulators (16 wmma / 16 ds_load per step)
//   double-buffered LDS fed by GLOBAL_LOAD_ASYNC_TO_LDS_B128
// FINAL=false : C stored bf16;  FINAL=true : relu(scaling*acc + bias) f32
// ---------------------------------------------------------------------------
template <bool FINAL>
__global__ __launch_bounds__(256) void gemm_bf16_kernel(
    const unsigned short* __restrict__ A,    // [MM][DD] bf16
    const unsigned short* __restrict__ Bt,   // [DD][DD] bf16 (N-major)
    void* __restrict__ Cout,
    const float* __restrict__ scaling,
    const float* __restrict__ bias) {
  constexpr int BM = 128, BN = 256, BK = 32, K = DD;

  __shared__ unsigned short sA[2][BM * BK];  // 2 x 8 KB
  __shared__ unsigned short sB[2][BN * BK];  // 2 x 16 KB  (48 KB total)

  const int tid      = threadIdx.x;
  const int lane     = tid & 31;
  const int wave     = tid >> 5;
  const int wm       = wave >> 2;            // 0..1  -> 64-row strips
  const int wn       = wave & 3;             // 0..3  -> 64-col strips
  const int lane15   = lane & 15;
  const int laneHalf = lane >> 4;            // 0/1 (K-half selector)
  const int bm       = blockIdx.y * BM;
  const int bn       = blockIdx.x * BN;

  v8f acc[4][4] = {};

  // stage one BK-slab of A and Bt into LDS buffer `buf`
  auto stage = [&](int buf, int kt) {
    #pragma unroll
    for (int c0 = 0; c0 < 2; ++c0) {         // A: 512 chunks of 8 bf16
      int c = tid + c0 * 256;
      int row = c >> 2, kc = (c & 3) << 3;
      copy16(&sA[buf][row * BK + kc], A + (size_t)(bm + row) * K + kt + kc);
    }
    #pragma unroll
    for (int c0 = 0; c0 < 4; ++c0) {         // B: 1024 chunks
      int c = tid + c0 * 256;
      int row = c >> 2, kc = (c & 3) << 3;
      copy16(&sB[buf][row * BK + kc], Bt + (size_t)(bn + row) * K + kt + kc);
    }
  };

  stage(0, 0);
  wait_async();
  __syncthreads();

  for (int kt = 0; kt < K; kt += BK) {
    const int cur = (kt / BK) & 1;
    if (kt + BK < K) stage(cur ^ 1, kt + BK);   // prefetch next slab (async)

    // ---- fragments ----
    union Frag { uint4 u[2]; v16bf v; };
    Frag af[4], bfg[4];
    #pragma unroll
    for (int mi = 0; mi < 4; ++mi) {
      const unsigned short* p =
          &sA[cur][(wm * 64 + mi * 16 + lane15) * BK + laneHalf * 8];
      af[mi].u[0] = *(const uint4*)(p);
      af[mi].u[1] = *(const uint4*)(p + 16);
    }
    #pragma unroll
    for (int ni = 0; ni < 4; ++ni) {
      const unsigned short* p =
          &sB[cur][(wn * 64 + ni * 16 + lane15) * BK + laneHalf * 8];
      bfg[ni].u[0] = *(const uint4*)(p);
      bfg[ni].u[1] = *(const uint4*)(p + 16);
    }
    #pragma unroll
    for (int mi = 0; mi < 4; ++mi)
      #pragma unroll
      for (int ni = 0; ni < 4; ++ni)
        acc[mi][ni] = __builtin_amdgcn_wmma_f32_16x16x32_bf16(
            false, af[mi].v, false, bfg[ni].v,
            (short)0, acc[mi][ni], false, false);

    wait_async();     // next slab landed (this wave's copies)...
    __syncthreads();  // ...and everyone else's; cur buffer free to overwrite
  }

  // ---- epilogue ----
  if (FINAL) {
    const float s = scaling[0];
    float* C = (float*)Cout;
    #pragma unroll
    for (int mi = 0; mi < 4; ++mi)
      #pragma unroll
      for (int ni = 0; ni < 4; ++ni) {
        const int n  = bn + wn * 64 + ni * 16 + lane15;
        const float bv = bias[n];
        #pragma unroll
        for (int r = 0; r < 8; ++r) {
          const int m = bm + wm * 64 + mi * 16 + laneHalf * 8 + r;
          float v = s * acc[mi][ni][r] + bv;
          C[(size_t)m * DD + n] = v > 0.f ? v : 0.f;
        }
      }
  } else {
    unsigned short* C = (unsigned short*)Cout;
    #pragma unroll
    for (int mi = 0; mi < 4; ++mi)
      #pragma unroll
      for (int ni = 0; ni < 4; ++ni) {
        const int n = bn + wn * 64 + ni * 16 + lane15;
        #pragma unroll
        for (int r = 0; r < 8; ++r) {
          const int m = bm + wm * 64 + mi * 16 + laneHalf * 8 + r;
          C[(size_t)m * DD + n] = f32_to_bf16_rne(acc[mi][ni][r]);
        }
      }
  }
}

// ---------------------------------------------------------------------------
extern "C" void kernel_launch(void* const* d_in, const int* in_sizes, int n_in,
                              void* d_out, int out_size, void* d_ws, size_t ws_size,
                              hipStream_t stream) {
  const float* x   = (const float*)d_in[0];
  const float* k0  = (const float*)d_in[1];
  const float* k1  = (const float*)d_in[2];
  const float* k2  = (const float*)d_in[3];
  const int*   m0  = (const int*)d_in[4];
  const int*   m1  = (const int*)d_in[5];
  const int*   m2  = (const int*)d_in[6];
  const float* scl = (const float*)d_in[7];
  const float* bia = (const float*)d_in[8];
  float*       out = (float*)d_out;

  char* ws = (char*)d_ws;
  const size_t MB = 1u << 20;
  unsigned short* Xb  = (unsigned short*)(ws + 0);         // 16 MB  [MM][DD] bf16
  unsigned short* W0t = (unsigned short*)(ws + 16 * MB);   //  8 MB
  unsigned short* W1t = (unsigned short*)(ws + 24 * MB);   //  8 MB
  unsigned short* W2t = (unsigned short*)(ws + 32 * MB);   //  8 MB
  unsigned short* Y1  = (unsigned short*)(ws + 40 * MB);   // 16 MB
  unsigned short* Y2  = Xb;                                // reuse X slot

  // convert inputs
  cvt_x_kernel<<<(MM * DD) / (256 * 4), 256, 0, stream>>>(x, Xb);
  {
    dim3 g(DD / 32, DD / 32), b(32, 8);
    cvt_w_kernel<<<g, b, 0, stream>>>(k0, m0, W0t);
    cvt_w_kernel<<<g, b, 0, stream>>>(k1, m1, W1t);
    cvt_w_kernel<<<g, b, 0, stream>>>(k2, m2, W2t);
  }

  // chained GEMMs: block tile 128x256
  dim3 gg(DD / 256, MM / 128);   // (8, 32)
  gemm_bf16_kernel<false><<<gg, 256, 0, stream>>>(Xb, W0t, (void*)Y1, nullptr, nullptr);
  gemm_bf16_kernel<false><<<gg, 256, 0, stream>>>(Y1, W1t, (void*)Y2, nullptr, nullptr);
  gemm_bf16_kernel<true ><<<gg, 256, 0, stream>>>(Y2, W2t, (void*)out, scl, bia);
}
